// NoiseAwareAttentionCABStyle_23914377904309
// MI455X (gfx1250) — matrix operand order repaired and checked by reading, not compiled, not run on hardware
//
#include <hip/hip_runtime.h>
#include <hip/hip_bf16.h>

typedef __attribute__((ext_vector_type(16))) __bf16         bf16x16;
typedef __attribute__((ext_vector_type(8)))  float          v8f;
typedef __attribute__((ext_vector_type(4)))  float          v4f;
typedef __attribute__((ext_vector_type(16))) unsigned short v16u16;
typedef __attribute__((ext_vector_type(8)))  unsigned short v8u16;
typedef __attribute__((ext_vector_type(4)))  unsigned short v4u16;

constexpr int HW    = 128 * 128;  // 16384
constexpr int CC    = 192;
constexpr int KC    = 6;          // 192 / 32 K-chunks
constexpr int HEADS = 6;
constexpr int BB    = 8;

// Native bf16 converts (backend emits hardware cvt; bf16->f32 is a shift).
__device__ __forceinline__ unsigned short f2bfu(float f) {
  return __builtin_bit_cast(unsigned short, (__bf16)f);
}
__device__ __forceinline__ float bf2f(unsigned short u) {
  return (float)__builtin_bit_cast(__bf16, u);
}

__device__ __forceinline__ v8f wmma_bf16(bf16x16 a, bf16x16 b, v8f c) {
  return __builtin_amdgcn_wmma_f32_16x16x32_bf16(
      false, a, false, b, (short)0, c, false, false);
}

// Build a bf16 fragment from two 8-element bf16 runs (16B vector loads).
__device__ __forceinline__ bf16x16 frag_from_bf_runs(
    const unsigned short* p0, const unsigned short* p1) {
  v8u16 a = *(const v8u16*)p0;
  v8u16 b = *(const v8u16*)p1;
  v16u16 r;
#pragma unroll
  for (int j = 0; j < 8; ++j) {
    r[j]     = a[j];
    r[8 + j] = b[j];
  }
  return __builtin_bit_cast(bf16x16, r);
}

// ---------------------------------------------------------------------------
// Pack a weight matrix W[OC x 192] (f32) into WMMA 16-bit A-fragment layout:
// dst[mt][kc][lane][i] : m = lane&15, kb = (lane&16)?8:0,
//                        k = kc*32 + (i<8 ? kb+i : kb+i+8)
// ---------------------------------------------------------------------------
__global__ __launch_bounds__(256) void pack_weight(
    const float* __restrict__ W, unsigned short* __restrict__ dst, int OC) {
  const int total = (OC >> 4) * KC * 32 * 16;
  int idx = blockIdx.x * 256 + threadIdx.x;
  if (idx >= total) return;
  int i = idx & 15;
  int rest = idx >> 4;
  int lane = rest & 31; rest >>= 5;
  int kc = rest % KC;
  int mt = rest / KC;
  int o  = mt * 16 + (lane & 15);
  int kb = (lane & 16) ? 8 : 0;
  int k  = kc * 32 + ((i < 8) ? (kb + i) : (kb + i + 8));
  dst[idx] = f2bfu(W[(size_t)o * CC + k]);
}

// ---------------------------------------------------------------------------
// GEMM: out[b][oc][n] = sum_ic A[oc][ic] * in[b][ic][n]   (IC = 192 fixed)
// IN_BF16 : activations are bf16 (else f32).  OUT_BF16 : store bf16 (else f32).
// GATED   : in *= sigmoid(ns_w[ic] * noise[b][n]).
// Block: 256 threads (8 waves), tile N=32, all OC; K staged through LDS.
// ---------------------------------------------------------------------------
template <bool GATED, bool IN_BF16, bool OUT_BF16>
__global__ __launch_bounds__(256) void gemm16(
    const unsigned short* __restrict__ Apack, long long aStrideB,
    const void* __restrict__ inv, long long inStrideB,
    void* __restrict__ outv, long long outStrideB, int OC,
    const float* __restrict__ ns_w, const float* __restrict__ noise) {
  __shared__ unsigned short sB[32 * 40];  // [n 0..31][k 0..31], stride 40 halfs
  const int b    = blockIdx.y;
  const int n0   = blockIdx.x << 5;
  const int t    = threadIdx.x;
  const int wave = t >> 5;
  const int lane = t & 31;
  const unsigned short* Ab = Apack + (long long)b * aStrideB;
  const int nsub = (OC >> 4) << 1;  // 16x16 subtiles (M-tiles x 2 N-tiles)

  const int sk = t >> 3;        // K row staged by this thread
  const int sn = (t & 7) << 2;  // first of 4 staged columns
  float g[4];
  if constexpr (GATED) {
    v4f nz = *(const v4f*)(noise + (size_t)b * HW + n0 + sn);
#pragma unroll
    for (int j = 0; j < 4; ++j) g[j] = nz[j];
  }

  v8f acc[6] = {};
  for (int kc = 0; kc < KC; ++kc) {
    const size_t soff = (size_t)(kc * 32 + sk) * HW + n0 + sn;
    unsigned short vals[4];
    if constexpr (IN_BF16) {
      const unsigned short* inB =
          (const unsigned short*)inv + (long long)b * inStrideB;
      v4u16 d = *(const v4u16*)(inB + soff);
      if constexpr (GATED) {
        float sc = ns_w[kc * 32 + sk];
#pragma unroll
        for (int j = 0; j < 4; ++j) {
          float f = bf2f(d[j]) * (1.f / (1.f + __expf(-sc * g[j])));
          vals[j] = f2bfu(f);
        }
      } else {
#pragma unroll
        for (int j = 0; j < 4; ++j) vals[j] = d[j];
      }
    } else {
      const float* inB = (const float*)inv + (long long)b * inStrideB;
      v4f d = *(const v4f*)(inB + soff);
      if constexpr (GATED) {
        float sc = ns_w[kc * 32 + sk];
#pragma unroll
        for (int j = 0; j < 4; ++j) d[j] *= 1.f / (1.f + __expf(-sc * g[j]));
      }
#pragma unroll
      for (int j = 0; j < 4; ++j) vals[j] = f2bfu(d[j]);
    }
#pragma unroll
    for (int j = 0; j < 4; ++j) sB[(sn + j) * 40 + sk] = vals[j];
    __syncthreads();

    int idx = 0;
    for (int s = wave; s < nsub; s += 8, ++idx) {
      const int mt = s >> 1, nt = s & 1;
      const v16u16* ap =
          (const v16u16*)(Ab + ((size_t)(mt * KC + kc) * 32 + lane) * 16);
      bf16x16 afrag = __builtin_bit_cast(bf16x16, *ap);
      const int nl   = (nt << 4) + (lane & 15);
      const int koff = (lane & 16) ? 8 : 0;  // 0 or 8
      bf16x16 bfrag = frag_from_bf_runs(&sB[nl * 40 + koff], &sB[nl * 40 + koff + 16]);
      acc[idx] = wmma_bf16(afrag, bfrag, acc[idx]);
    }
    __syncthreads();
  }

  int idx = 0;
  for (int s = wave; s < nsub; s += 8, ++idx) {
    const int mt = s >> 1, nt = s & 1;
    const int n     = n0 + (nt << 4) + (lane & 15);
    const int mbase = mt * 16 + ((lane & 16) ? 8 : 0);
    if constexpr (OUT_BF16) {
      unsigned short* outB =
          (unsigned short*)outv + (long long)b * outStrideB;
#pragma unroll
      for (int r = 0; r < 8; ++r)
        outB[(size_t)(mbase + r) * HW + n] = f2bfu(acc[idx][r]);
    } else {
      float* outB = (float*)outv + (long long)b * outStrideB;
#pragma unroll
      for (int r = 0; r < 8; ++r)
        outB[(size_t)(mbase + r) * HW + n] = acc[idx][r];
    }
  }
}

// ---------------------------------------------------------------------------
// Depthwise 3x3, SAME zero padding. bf16 in/out, f32 math. wt [Ctot][9] f32.
// ---------------------------------------------------------------------------
__global__ __launch_bounds__(256) void dwconv3(
    const unsigned short* __restrict__ in, const float* __restrict__ wt,
    unsigned short* __restrict__ out, int Ctot) {
  const int n = blockIdx.x * 256 + threadIdx.x;
  const int c = blockIdx.y;
  const int b = blockIdx.z;
  const unsigned short* ip = in + ((size_t)b * Ctot + c) * HW;
  const float* w9 = wt + (size_t)c * 9;
  const int hh = n >> 7, ww = n & 127;
  float s = 0.f;
#pragma unroll
  for (int dy = 0; dy < 3; ++dy) {
    int y = hh + dy - 1;
    if ((unsigned)y >= 128u) continue;
#pragma unroll
    for (int dx = 0; dx < 3; ++dx) {
      int x = ww + dx - 1;
      if ((unsigned)x >= 128u) continue;
      s += w9[dy * 3 + dx] * bf2f(ip[y * 128 + x]);
    }
  }
  out[((size_t)b * Ctot + c) * HW + n] = f2bfu(s);
}

// ---------------------------------------------------------------------------
// attn logits split-K partials: part[split][bh][32][32] = q . k^T over a
// 1024-wide N slice. bf16 inputs feed WMMA directly (no conversion).
// Deterministic: fixed-order reductions, no atomics.
// ---------------------------------------------------------------------------
__global__ __launch_bounds__(256) void attn_partial(
    const unsigned short* __restrict__ qf, const unsigned short* __restrict__ kvf,
    float* __restrict__ part) {
  __shared__ float sAcc[8 * 1024];
  const int bh = blockIdx.x, b = bh / HEADS, h = bh % HEADS;
  const int t = threadIdx.x, wave = t >> 5, lane = t & 31;
  const unsigned short* qb = qf + ((size_t)b * CC + h * 32) * HW;
  const unsigned short* kb = kvf + ((size_t)b * 2 * CC + h * 32) * HW;
  v8f acc[4] = {};
  const int nstart = (blockIdx.y * 8 + wave) * 128;
  const int kbA = (lane & 16) ? 8 : 0;
  const int kbB = (lane & 16) ? 8 : 0;
  const int row = lane & 15;
  for (int it = 0; it < 4; ++it) {
    const int nb = nstart + it * 32;
    const unsigned short* pa0 = qb + (size_t)row * HW + nb + kbA;
    const unsigned short* pa1 = qb + (size_t)(16 + row) * HW + nb + kbA;
    bf16x16 A0 = frag_from_bf_runs(pa0, pa0 + 16);
    bf16x16 A1 = frag_from_bf_runs(pa1, pa1 + 16);
    const unsigned short* pb0 = kb + (size_t)row * HW + nb + kbB;
    const unsigned short* pb1 = kb + (size_t)(16 + row) * HW + nb + kbB;
    bf16x16 B0 = frag_from_bf_runs(pb0, pb0 + 16);
    bf16x16 B1 = frag_from_bf_runs(pb1, pb1 + 16);
    acc[0] = wmma_bf16(A0, B0, acc[0]);
    acc[1] = wmma_bf16(A0, B1, acc[1]);
    acc[2] = wmma_bf16(A1, B0, acc[2]);
    acc[3] = wmma_bf16(A1, B1, acc[3]);
  }
  const int mo = (lane & 16) ? 8 : 0;
#pragma unroll
  for (int st = 0; st < 4; ++st) {
    const int mt = st >> 1, nt = st & 1;
#pragma unroll
    for (int r = 0; r < 8; ++r) {
      int cc = mt * 16 + r + mo;
      int dd = nt * 16 + row;
      sAcc[wave * 1024 + cc * 32 + dd] = acc[st][r];
    }
  }
  __syncthreads();
  for (int i = t; i < 1024; i += 256) {
    float s = 0.f;
#pragma unroll
    for (int w2 = 0; w2 < 8; ++w2) s += sAcc[w2 * 1024 + i];
    part[((size_t)blockIdx.y * (BB * HEADS) + bh) * 1024 + i] = s;
  }
}

// ---------------------------------------------------------------------------
// Reduce 16 split-K partials (fixed order), apply temperature, row softmax.
// Block (32,32): one wave per row of the 32x32 attention tile.
// ---------------------------------------------------------------------------
__global__ void attn_softmax(const float* __restrict__ part,
                             float* __restrict__ soft,
                             const float* __restrict__ temperature) {
  const int bh = blockIdx.x, h = bh % HEADS;
  const int c = threadIdx.x, r = threadIdx.y;
  float v = 0.f;
  for (int s = 0; s < 16; ++s)
    v += part[((size_t)s * (BB * HEADS) + bh) * 1024 + r * 32 + c];
  v *= temperature[h];
  float mx = v;
  for (int m = 16; m > 0; m >>= 1) mx = fmaxf(mx, __shfl_xor(mx, m, 32));
  float e = __expf(v - mx);
  float sum = e;
  for (int m = 16; m > 0; m >>= 1) sum += __shfl_xor(sum, m, 32);
  soft[(size_t)bh * 1024 + r * 32 + c] = e / sum;
}

// ---------------------------------------------------------------------------
// M_b = proj_w . blockdiag(attn_b), packed directly into bf16 A-fragment
// layout for the final GEMM: Mpack[b][mt][kc][lane][i].
// ---------------------------------------------------------------------------
__global__ __launch_bounds__(256) void combine_pack(
    const float* __restrict__ proj_w, const float* __restrict__ soft,
    unsigned short* __restrict__ Mp) {
  const int total = BB * 12 * KC * 32 * 16;  // 294912
  int idx = blockIdx.x * 256 + threadIdx.x;
  if (idx >= total) return;
  int i = idx & 15;
  int rest = idx >> 4;
  int lane = rest & 31; rest >>= 5;
  int kc = rest % KC; rest /= KC;
  int mt = rest % 12;
  int b  = rest / 12;
  int o  = mt * 16 + (lane & 15);
  int kb = (lane & 16) ? 8 : 0;
  int k  = kc * 32 + ((i < 8) ? (kb + i) : (kb + i + 8));
  int h  = k >> 5;
  int d  = k & 31;
  const float* pw = proj_w + (size_t)o * CC + h * 32;
  const float* at = soft + (size_t)(b * HEADS + h) * 1024 + d;
  float s = 0.f;
#pragma unroll
  for (int c = 0; c < 32; ++c) s += pw[c] * at[c * 32];
  Mp[idx] = f2bfu(s);
}

// ---------------------------------------------------------------------------
// Workspace layout (bytes), bf16 intermediates. q_pre aliases kv_feat
// (q_pre dead before dw-kv overwrites it). Peak ~256 MiB.
// ---------------------------------------------------------------------------
constexpr size_t SZ_KVH     = (size_t)BB * 2 * CC * HW * 2;  // 100663296
constexpr size_t SZ_QH      = (size_t)BB * CC * HW * 2;      // 50331648
constexpr size_t OFF_KVFEAT = 0;                     // kv_feat (and q_pre alias)
constexpr size_t OFF_QFEAT  = SZ_KVH;                // q_feat
constexpr size_t OFF_KVPRE  = SZ_KVH + SZ_QH;        // kv_pre
constexpr size_t OFF_PART   = SZ_KVH + SZ_QH + SZ_KVH;
constexpr size_t OFF_SOFT   = OFF_PART + (size_t)16 * 48 * 1024 * 4;
constexpr size_t OFF_APQ    = OFF_SOFT + (size_t)48 * 1024 * 4;
constexpr size_t OFF_APKV   = OFF_APQ + (size_t)12 * KC * 32 * 16 * 2;
constexpr size_t OFF_MP     = OFF_APKV + (size_t)24 * KC * 32 * 16 * 2;

extern "C" void kernel_launch(void* const* d_in, const int* in_sizes, int n_in,
                              void* d_out, int out_size, void* d_ws,
                              size_t ws_size, hipStream_t stream) {
  (void)in_sizes; (void)n_in; (void)out_size; (void)ws_size;
  const float* x      = (const float*)d_in[0];
  const float* y      = (const float*)d_in[1];
  const float* noise  = (const float*)d_in[2];
  const float* q_w    = (const float*)d_in[3];
  const float* q_dw   = (const float*)d_in[4];
  const float* kv_w   = (const float*)d_in[5];
  const float* kv_dw  = (const float*)d_in[6];
  const float* ns_w   = (const float*)d_in[7];
  const float* proj_w = (const float*)d_in[8];
  const float* temp   = (const float*)d_in[9];
  float* out = (float*)d_out;

  char* ws = (char*)d_ws;
  unsigned short* kv_feat = (unsigned short*)(ws + OFF_KVFEAT);
  unsigned short* q_pre   = kv_feat;  // alias: dead before dw-kv overwrites it
  unsigned short* q_feat  = (unsigned short*)(ws + OFF_QFEAT);
  unsigned short* kv_pre  = (unsigned short*)(ws + OFF_KVPRE);
  float* part             = (float*)(ws + OFF_PART);
  float* soft             = (float*)(ws + OFF_SOFT);
  unsigned short* ApQ     = (unsigned short*)(ws + OFF_APQ);
  unsigned short* ApKV    = (unsigned short*)(ws + OFF_APKV);
  unsigned short* Mp      = (unsigned short*)(ws + OFF_MP);

  const long long sQ  = (long long)CC * HW;       // per-batch C-plane stride
  const long long sKV = (long long)2 * CC * HW;

  // 1) Pack 1x1-conv weights into WMMA A-fragment layout (bf16).
  pack_weight<<<144, 256, 0, stream>>>(q_w, ApQ, CC);
  pack_weight<<<288, 256, 0, stream>>>(kv_w, ApKV, 2 * CC);

  // 2) q_pre = q_w @ x ; kv_pre = kv_w @ y   (f32 in, bf16 out, WMMA)
  gemm16<false, false, true><<<dim3(HW / 32, BB), 256, 0, stream>>>(
      ApQ, 0, x, sQ, q_pre, sQ, CC, nullptr, nullptr);
  // 3) q_feat = dwconv3(q_pre)   (bf16 -> bf16, f32 math)
  dwconv3<<<dim3(HW / 256, CC, BB), 256, 0, stream>>>(q_pre, q_dw, q_feat, CC);
  gemm16<false, false, true><<<dim3(HW / 32, BB), 256, 0, stream>>>(
      ApKV, 0, y, sQ, kv_pre, sKV, 2 * CC, nullptr, nullptr);
  // 4) kv_feat = dwconv3(kv_pre)   (overwrites dead q_pre alias)
  dwconv3<<<dim3(HW / 256, 2 * CC, BB), 256, 0, stream>>>(kv_pre, kv_dw,
                                                          kv_feat, 2 * CC);

  // 5) attn logits (split-K WMMA partials), then reduce + softmax.
  attn_partial<<<dim3(BB * HEADS, 16), 256, 0, stream>>>(q_feat, kv_feat, part);
  attn_softmax<<<BB * HEADS, dim3(32, 32), 0, stream>>>(part, soft, temp);

  // 6) M_b = proj_w . blockdiag(attn_b), packed for WMMA.
  combine_pack<<<1152, 256, 0, stream>>>(proj_w, soft, Mp);

  // 7) final = M_b @ (v * sigmoid(ns_w * noise))   (bf16 in, f32 out) -> d_out
  gemm16<true, true, false><<<dim3(HW / 32, BB), 256, 0, stream>>>(
      Mp, (long long)12 * KC * 32 * 16, kv_feat + (size_t)CC * HW, sKV, out,
      sQ, CC, ns_w, noise);
}
